// TransConv_74380243632580
// MI455X (gfx1250) — matrix operand order repaired and checked by reading, not compiled, not run on hardware
//
#include <hip/hip_runtime.h>

typedef _Float16 f16;
typedef __attribute__((ext_vector_type(8)))  _Float16 v8h;
typedef __attribute__((ext_vector_type(16))) _Float16 v16h;
typedef __attribute__((ext_vector_type(8)))  float    v8f;
typedef __attribute__((ext_vector_type(4)))  int      v4i;

typedef __attribute__((address_space(1))) v4i g_v4i;   // global int4
typedef __attribute__((address_space(3))) v4i l_v4i;   // LDS int4

#define OC 128
#define LROW 40          // 32 halves + 8 pad per LDS row (80B, keeps 16B align)
#define HW 34            // halo-padded spatial dim
#define HWW (HW*HW)      // 1156

#if __has_builtin(__builtin_amdgcn_global_load_async_to_lds_b128) && \
    __has_builtin(__builtin_amdgcn_s_wait_asynccnt)
#define USE_ASYNC 1
#else
#define USE_ASYNC 0
#endif

__device__ __forceinline__ void copy16(const f16* __restrict__ src,
                                       f16* __restrict__ dst_lds) {
#if USE_ASYNC
    __builtin_amdgcn_global_load_async_to_lds_b128(
        (g_v4i*)src, (l_v4i*)dst_lds, 0, 0);
#else
    *(v8h*)dst_lds = *(const v8h*)src;
#endif
}

__device__ __forceinline__ void wait_async() {
#if USE_ASYNC
    __builtin_amdgcn_s_wait_asynccnt(0);
#endif
}

// -------- prologue 1: W (OIHW f32) -> Wh2[pc][oc][tap*256+ic] f16 -----------
// kh = py + 2*dh, kw = px + 2*dw ; pc = py*2+px ; tap = dh*2+dw
__global__ __launch_bounds__(256) void cvt_w(const float* __restrict__ W,
                                             f16* __restrict__ Wh2) {
    int idx = blockIdx.x * 256 + threadIdx.x;      // 524288
    int kw = idx & 3;
    int kh = (idx >> 2) & 3;
    int ic = (idx >> 4) & 255;
    int oc = idx >> 12;
    int py = kh & 1, dh = kh >> 1;
    int px = kw & 1, dw = kw >> 1;
    int pc = py * 2 + px;
    int tap = dh * 2 + dw;
    Wh2[((pc * OC + oc) << 10) + (tap << 8) + ic] = (f16)W[idx];
}

// -------- prologue 2: x NCHW f32 -> xh halo-padded NHWC f16 [32][34][34][256]
__global__ __launch_bounds__(256) void cvt_x(const float* __restrict__ x,
                                             f16* __restrict__ xh) {
    int idx = blockIdx.x * 256 + threadIdx.x;      // 32*34*34*256 = 9469952
    int ic = idx & 255;
    int t  = idx >> 8;            // n*1156 + iyp*34 + ixp
    int ixp = t % HW;
    int tt  = t / HW;
    int iyp = tt % HW;
    int n   = tt / HW;
    int iy = iyp - 1, ix = ixp - 1;
    f16 v = (f16)0;
    if ((unsigned)iy < 32u && (unsigned)ix < 32u)
        v = (f16)x[(((n << 8) + ic) << 10) + (iy << 5) + ix];
    xh[idx] = v;
}

// -------- main: per-parity implicit GEMM, double-buffered async staging -----
// grid = (rg:4, n:32, pc:4), block = 256 (8 wave32)
// block tile: M=128 (all oc) x N=256 positions (8 output rows of one parity)
// wave tile : M=128 x N=32  -> 16 x v_wmma_f32_16x16x32_f16 per K-step
__global__ __launch_bounds__(256) void tconv_wmma(const f16* __restrict__ Wh2,
                                                  const f16* __restrict__ xh,
                                                  const float* __restrict__ bias,
                                                  float* __restrict__ out) {
    __shared__ f16 lA[2][OC * LROW];      // 2 x 10240 B
    __shared__ f16 lB[2][256 * LROW];     // 2 x 20480 B

    const int tid = threadIdx.x;
    const int rg  = blockIdx.x;           // 8 output rows per group
    const int n   = blockIdx.y;
    const int pc  = blockIdx.z;
    const int py  = pc >> 1, px = pc & 1;
    const int r0  = rg << 3;

    const int wv   = tid >> 5;
    const int lane = tid & 31;
    const int lh   = lane >> 4;
    const int ln   = lane & 15;

    v8f acc[2][8] = {};

    const f16* wbase = Wh2 + ((size_t)pc << 17);

    // ---- precompute per-thread staging chunk addresses (kb-invariant parts)
    int apre[2], adst[2];
    #pragma unroll
    for (int i = 0; i < 2; ++i) {
        int ch = tid + (i << 8);
        int oc = ch >> 2;
        int ko = (ch & 3) << 3;
        apre[i] = (oc << 10) + ko;
        adst[i] = oc * LROW + ko;
    }
    int bpre[4], bdst[4];
    #pragma unroll
    for (int i = 0; i < 4; ++i) {
        int ch  = tid + (i << 8);
        int pos = ch >> 2;                 // 0..255
        int ko  = (ch & 3) << 3;
        int rr  = pos >> 5;                // 0..7
        int c   = pos & 31;
        bpre[i] = ((n * HWW + (r0 + rr + py) * HW + (c + px)) << 8) + ko;
        bdst[i] = pos * LROW + ko;
    }

    // ---- stage tile kb into buffer bi (tap & ic-block offsets are uniform)
    auto stage = [&](int kb, int bi) {
        const int tap  = kb >> 3;
        const int dh   = tap >> 1, dw = tap & 1;
        const int aoff = kb << 5;
        const int boff = ((dh * HW + dw) << 8) + ((kb & 7) << 5);
        #pragma unroll
        for (int i = 0; i < 2; ++i)
            copy16(wbase + apre[i] + aoff, &lA[bi][adst[i]]);
        #pragma unroll
        for (int i = 0; i < 4; ++i)
            copy16(xh + bpre[i] + boff, &lB[bi][bdst[i]]);
    };

    stage(0, 0);

    #pragma unroll 2
    for (int kb = 0; kb < 32; ++kb) {
        const int bi = kb & 1;
        wait_async();                      // my async copies for tile kb done
        __syncthreads();                   // tile kb visible to all waves
        if (kb + 1 < 32) stage(kb + 1, bi ^ 1);   // overlap with compute

        // ---- two B fragments (wave strips wv and wv+8)
        v16h bf[2];
        #pragma unroll
        for (int t = 0; t < 2; ++t) {
            const f16* bptr = &lB[bi][(((wv + (t << 3)) << 4) + ln) * LROW + (lh << 4)];
            v8h blo = *(const v8h*)(bptr);
            v8h bhi = *(const v8h*)(bptr + 8);
            bf[t] = __builtin_shufflevector(blo, bhi,
                        0,1,2,3,4,5,6,7,8,9,10,11,12,13,14,15);
        }
        // ---- 8 A fragments, each feeding two WMMAs
        #pragma unroll
        for (int m = 0; m < 8; ++m) {
            const f16* aptr = &lA[bi][((m << 4) + ln) * LROW + (lh << 4)];
            v8h alo = *(const v8h*)(aptr);
            v8h ahi = *(const v8h*)(aptr + 8);
            v16h af = __builtin_shufflevector(alo, ahi,
                        0,1,2,3,4,5,6,7,8,9,10,11,12,13,14,15);
            acc[0][m] = __builtin_amdgcn_wmma_f32_16x16x32_f16(
                            false, af, false, bf[0], (short)0, acc[0][m], false, false);
            acc[1][m] = __builtin_amdgcn_wmma_f32_16x16x32_f16(
                            false, af, false, bf[1], (short)0, acc[1][m], false, false);
        }
    }

    // ---- epilogue: C layout -> NCHW output, add bias
    const int cc = ((wv & 1) << 4) + ln;
    const int ox = (cc << 1) + px;
    #pragma unroll
    for (int t = 0; t < 2; ++t) {
        const int rr = (wv >> 1) + (t << 2);
        const int oy = ((r0 + rr) << 1) + py;
        #pragma unroll
        for (int m = 0; m < 8; ++m) {
            #pragma unroll
            for (int v = 0; v < 8; ++v) {
                int oc = (m << 4) + v + (lh << 3);
                out[(((size_t)n * OC + oc) << 12) + (oy << 6) + ox] =
                    acc[t][m][v] + bias[oc];
            }
        }
    }
}

extern "C" void kernel_launch(void* const* d_in, const int* in_sizes, int n_in,
                              void* d_out, int out_size, void* d_ws, size_t ws_size,
                              hipStream_t stream) {
    (void)in_sizes; (void)n_in; (void)out_size; (void)ws_size;
    const float* x = (const float*)d_in[0];   // (32,256,32,32)
    const float* W = (const float*)d_in[1];   // (128,256,4,4)
    const float* b = (const float*)d_in[2];   // (128,)
    float* out = (float*)d_out;               // (32,128,64,64)

    f16* Wh2 = (f16*)d_ws;                              // 1 MB
    f16* xh  = (f16*)((char*)d_ws + (1u << 20));        // 18.1 MB (halo NHWC)

    cvt_w<<<524288 / 256, 256, 0, stream>>>(W, Wh2);
    cvt_x<<<(32 * HWW * 256) / 256, 256, 0, stream>>>(x, xh);

    dim3 grid(4, 32, 4);
    tconv_wmma<<<grid, 256, 0, stream>>>(Wh2, xh, b, out);
}
